// RGCN_61194694034310
// MI455X (gfx1250) — matrix operand order repaired and checked by reading, not compiled, not run on hardware
//
#include <hip/hip_runtime.h>

#define NN 50000
#define EE 800000
#define DD 128
#define RR 8
#define BBASES 4
#define KTOT (RR * DD + DD)   // 1152 (8 relations + root folded in)
#define NKT (KTOT / 32)       // 36 K-tiles of 32
#define NKP (NKT / 2)         // 18 double K-tiles of 64

typedef __attribute__((ext_vector_type(16))) __bf16 v16bf;
typedef __attribute__((ext_vector_type(8)))  float  v8f;

// f32x2 -> packed bf16x2 (round-half-up), 3 VALU ops via v_perm_b32
__device__ __forceinline__ unsigned pack2_bf16(float a, float b) {
    unsigned ua = __float_as_uint(a) + 0x8000u;
    unsigned ub = __float_as_uint(b) + 0x8000u;
    // result bytes (LSB..MSB): ua.b2, ua.b3, ub.b2, ub.b3  -> (bf16(b)<<16)|bf16(a)
    return __builtin_amdgcn_perm(ub, ua, 0x07060302u);
}

// ---------------------------------------------------------------- x0 = emb[entity]
__global__ void gather_x_kernel(const int* __restrict__ entity,
                                const float* __restrict__ emb,
                                float* __restrict__ x0) {
    int tid = blockIdx.x * blockDim.x + threadIdx.x;   // NN*32 threads
    int n = tid >> 5, q = tid & 31;
    if (n >= NN) return;
    const float4* s = (const float4*)(emb + (size_t)entity[n] * DD);
    float4* d = (float4*)(x0 + (size_t)n * DD);
    d[q] = s[q];
}

// ------------------------------------------- pack [W_0..W_7 ; root] (bf16) into the
// per-lane B-fragment layout this file's GEMM uses (contiguous k): dword index = ((kt*8 + ct)*32 + lane)*8 + j
// lane<16: col = ct*16+lane, K = 2j,2j+1 ; lane>=16: col = ct*16+lane-16, K = 16+2j,...
__global__ void pack_w_kernel(const float* __restrict__ comp,    // [R,B]
                              const float* __restrict__ bases,   // [B,D,D]
                              const float* __restrict__ root,    // [D,D]
                              unsigned* __restrict__ wpack) {
    int gid = blockIdx.x * blockDim.x + threadIdx.x;             // 36*8*32*8 = 73728
    if (gid >= NKT * 8 * 32 * 8) return;
    int j    = gid & 7;
    int lane = (gid >> 3) & 31;
    int ct   = (gid >> 8) & 7;
    int kt   = gid >> 11;
    int col  = ct * 16 + (lane & 15);
    int k    = kt * 32 + ((lane >> 4) << 4) + 2 * j;
    float v0, v1;
    if (k < RR * DD) {
        int r = k >> 7, i = k & 127;                  // W[r][i][col] = sum_b comp*bases
        float a0 = 0.f, a1 = 0.f;
        #pragma unroll
        for (int b = 0; b < BBASES; ++b) {
            float cm = comp[r * BBASES + b];
            a0 += cm * bases[((size_t)b * DD + i    ) * DD + col];
            a1 += cm * bases[((size_t)b * DD + i + 1) * DD + col];
        }
        v0 = a0; v1 = a1;
    } else {
        int i = k - RR * DD;
        v0 = root[(size_t)i * DD + col];
        v1 = root[(size_t)(i + 1) * DD + col];
    }
    wpack[gid] = pack2_bf16(v0, v1);
}

// ------------------------------------------- edge scatter: H[dst,rt,:] += x[src,:]
__global__ void scatter_kernel(const float* __restrict__ x,
                               const int* __restrict__ ei,    // [2,E] flat
                               const int* __restrict__ et,    // [E]
                               float* __restrict__ H,         // [N*R, D]
                               float* __restrict__ cnt) {     // [N*R]
    int wave = threadIdx.x >> 5, lane = threadIdx.x & 31;
    int e = blockIdx.x * 8 + wave;
    if (e >= EE) return;
    int src = ei[e], dst = ei[EE + e], rt = et[e];
    size_t seg = (size_t)dst * RR + rt;
    const float4 v = ((const float4*)(x + (size_t)src * DD))[lane];
    float* hp = H + seg * DD + (size_t)lane * 4;
    unsafeAtomicAdd(hp + 0, v.x);
    unsafeAtomicAdd(hp + 1, v.y);
    unsafeAtomicAdd(hp + 2, v.z);
    unsafeAtomicAdd(hp + 3, v.w);
    if (lane == 0) unsafeAtomicAdd(cnt + seg, 1.0f);
}

// ------------------------------------------- fused mean + GEMM + bias (+ReLU)
// out[n0+m, :] = sum_kt A(kt) @ B(kt),  A = [H/cnt (kt<32) | x (kt>=32)] in bf16
// block = 256 threads = 8 waves; wave w -> output columns w*16..w*16+15
// K=64 per iteration (two 16x16x32 WMMAs per wave per barrier); both sub-tiles of an
// iteration share one relation r, so the 64 staged k-values are one contiguous chunk.
__launch_bounds__(256)
__global__ void rgcn_gemm_kernel(const float* __restrict__ H,
                                 const float* __restrict__ cnt,
                                 const float* __restrict__ x,
                                 const unsigned* __restrict__ wpack,
                                 const float* __restrict__ bias,
                                 float* __restrict__ out,
                                 int relu) {
    __shared__ alignas(16) unsigned char sA[2][16 * 144];  // 16 rows, 64 bf16 + 16B pad
    const int t    = threadIdx.x;
    const int wv   = t >> 5;
    const int lane = t & 31;
    const int n0   = blockIdx.x * 16;
    const int srow = t >> 4;          // staging: row 0..15
    const int skp  = t & 15;          // staging: float4 index 0..15 (k = 4*skp..4*skp+3)

    v8f c = {};

    auto stage = [&](int ktp, int buf) {
        int n = n0 + srow;
        const float* p; float s;
        if (ktp < 16) {                               // relation part, mean-scaled
            int r = ktp >> 1;
            size_t seg = (size_t)n * RR + r;
            p = H + seg * DD + (ktp & 1) * 64 + skp * 4;
            s = __builtin_amdgcn_rcpf(fmaxf(cnt[seg], 1.0f));
        } else {                                      // root part: raw x
            p = x + (size_t)n * DD + (ktp - 16) * 64 + skp * 4;
            s = 1.0f;
        }
        const float4 v = *(const float4*)p;
        uint2 pk;
        pk.x = pack2_bf16(v.x * s, v.y * s);
        pk.y = pack2_bf16(v.z * s, v.w * s);
        *(uint2*)(&sA[buf][srow * 144 + skp * 8]) = pk;
    };

    stage(0, 0);
    __syncthreads();

    for (int ktp = 0; ktp < NKP; ++ktp) {
        int buf = ktp & 1;
        if (ktp + 1 < NKP) stage(ktp + 1, buf ^ 1);

        // A fragments from LDS: lane<16 -> row=lane, K 0..15 ; lane>=16 -> K 16..31
        const unsigned char* rb = &sA[buf][(lane & 15) * 144 + ((lane >> 4) << 5)];
        if (ktp + 1 < NKP)
            __builtin_prefetch(wpack + (((size_t)(2 * ktp + 2) * 8 + wv) * 32 + lane) * 8, 0, 3);
        #pragma unroll
        for (int s2 = 0; s2 < 2; ++s2) {
            int kt = 2 * ktp + s2;
            union { uint4 u[2]; v16bf v; } af, bfr;
            af.u[0] = *(const uint4*)(rb + s2 * 64);
            af.u[1] = *(const uint4*)(rb + s2 * 64 + 16);
            const uint4* bp = (const uint4*)(wpack + (((size_t)kt * 8 + wv) * 32 + lane) * 8);
            bfr.u[0] = bp[0];
            bfr.u[1] = bp[1];
            c = __builtin_amdgcn_wmma_f32_16x16x32_bf16(false, af.v, false, bfr.v,
                                                        (short)0, c, false, false);
        }
        __syncthreads();
    }

    // epilogue: C layout — VGPR i: row = i (lanes 0-15) / i+8 (lanes 16-31), col = lane%16
    int col   = wv * 16 + (lane & 15);
    int rbase = n0 + ((lane >> 4) << 3);
    float bb  = bias[col];
    #pragma unroll
    for (int i = 0; i < 8; ++i) {
        float v = c[i] + bb;
        if (relu) v = fmaxf(v, 0.0f);
        out[(size_t)(rbase + i) * DD + col] = v;
    }
}

extern "C" void kernel_launch(void* const* d_in, const int* in_sizes, int n_in,
                              void* d_out, int out_size, void* d_ws, size_t ws_size,
                              hipStream_t stream) {
    const int*   entity = (const int*)d_in[0];
    const int*   ei     = (const int*)d_in[1];
    const int*   et     = (const int*)d_in[2];
    const float* emb    = (const float*)d_in[3];
    const float* comp1  = (const float*)d_in[4];
    const float* bases1 = (const float*)d_in[5];
    const float* root1  = (const float*)d_in[6];
    const float* bias1  = (const float*)d_in[7];
    const float* comp2  = (const float*)d_in[8];
    const float* bases2 = (const float*)d_in[9];
    const float* root2  = (const float*)d_in[10];
    const float* bias2  = (const float*)d_in[11];

    char* ws = (char*)d_ws;
    size_t off = 0;
    float* H    = (float*)(ws + off); off += (size_t)NN * RR * DD * 4;   // 204.8 MB
    float* cnt  = (float*)(ws + off); off += (size_t)NN * RR * 4;        // 1.6 MB
    float* x0   = (float*)(ws + off); off += (size_t)NN * DD * 4;        // 25.6 MB
    float* x1   = (float*)(ws + off); off += (size_t)NN * DD * 4;        // 25.6 MB
    unsigned* wpack = (unsigned*)(ws + off); off += (size_t)NKT * 8 * 32 * 8 * 4; // 0.3 MB

    const dim3 B256(256);
    const int packGrid = (NKT * 8 * 32 * 8 + 255) / 256;

    // ---------------- layer 1 ----------------
    gather_x_kernel<<<dim3((NN * 32 + 255) / 256), B256, 0, stream>>>(entity, emb, x0);
    pack_w_kernel<<<dim3(packGrid), B256, 0, stream>>>(comp1, bases1, root1, wpack);
    hipMemsetAsync(H,   0, (size_t)NN * RR * DD * 4, stream);
    hipMemsetAsync(cnt, 0, (size_t)NN * RR * 4, stream);
    scatter_kernel<<<dim3((EE + 7) / 8), B256, 0, stream>>>(x0, ei, et, H, cnt);
    rgcn_gemm_kernel<<<dim3(NN / 16), B256, 0, stream>>>(H, cnt, x0, wpack, bias1, x1, 1);

    // ---------------- layer 2 ----------------
    pack_w_kernel<<<dim3(packGrid), B256, 0, stream>>>(comp2, bases2, root2, wpack);
    hipMemsetAsync(H,   0, (size_t)NN * RR * DD * 4, stream);
    hipMemsetAsync(cnt, 0, (size_t)NN * RR * 4, stream);
    scatter_kernel<<<dim3((EE + 7) / 8), B256, 0, stream>>>(x1, ei, et, H, cnt);
    rgcn_gemm_kernel<<<dim3(NN / 16), B256, 0, stream>>>(H, cnt, x1, wpack, bias2,
                                                         (float*)d_out, 0);
}
